// ModulatedConv2d_70720931496590
// MI455X (gfx1250) — matrix operand order repaired and checked by reading, not compiled, not run on hardware
//
#include <hip/hip_runtime.h>
#include <hip/hip_bf16.h>

typedef __bf16 bf16_t;
typedef bf16_t v16bf __attribute__((ext_vector_type(16)));
typedef bf16_t v8bf  __attribute__((ext_vector_type(8)));
typedef float  v8f   __attribute__((ext_vector_type(8)));
typedef unsigned int u32;
typedef u32 v4u __attribute__((ext_vector_type(4)));
typedef int  v8i __attribute__((ext_vector_type(8)));
typedef int  v4i __attribute__((ext_vector_type(4)));

#define BATCH 16
#define IN_C  512
#define OUT_C 512
#define WDIM  512
#define IMG   32
#define HW    1024          // 32*32 pixels per image
// 1/sqrt(512*9), 1/sqrt(512)
#define CONV_SCALE  0.014731391274719739f
#define STYLE_SCALE 0.044194173824159216f

#define LX_ELEMS (6 * 34 * 32)   // [6 padded rows][34 padded cols][32 channels]

__device__ __forceinline__ unsigned short f2bf(float f) {
    unsigned int u = __float_as_uint(f);
    unsigned int r = (u + 0x7FFFu + ((u >> 16) & 1u)) >> 16;   // RNE
    return (unsigned short)r;
}

// ---------------------------------------------------------------------------
// TDM: 2D tile load Global->LDS via tensor_load_to_lds (6-arg toolchain form:
// g0 v4u, g1 v8i, g2 v4i, g3 v4i, extra v8i, cpol i32).
// Tile: 128 rows x 32 bf16 elems, row stride 512 elems, contiguous in LDS.
// D# layout per CDNA5 ISA section 8.3/8.4.
// ---------------------------------------------------------------------------
__device__ __forceinline__ void tdm_load_tileA(u32 lds_off,
                                               const unsigned short* gptr)
{
    unsigned long long ga = (unsigned long long)(uintptr_t)gptr;
    v4u g0;
    g0[0] = 1u;                                   // count=1, user descriptor
    g0[1] = lds_off;                              // LDS byte address
    g0[2] = (u32)ga;                              // global_addr[31:0]
    g0[3] = ((u32)(ga >> 32) & 0x01FFFFFFu) | (2u << 30);  // addr[56:32], type=2

    const u32 tensor_w = 1u << 20;                // generous (tile in-bounds)
    const u32 tensor_h = 1u << 20;
    const u32 tile_w   = 32u;                     // elems (64B rows)
    const u32 tile_h   = 128u;
    const u32 stride0  = (u32)IN_C;               // elems between rows

    v8i g1;
    g1[0] = (int)(1u << 16);                      // wg_mask=0, data_size=1 (2B)
    g1[1] = (int)((tensor_w & 0xFFFFu) << 16);    // tensor_dim0[15:0] @bit48
    g1[2] = (int)(((tensor_w >> 16) & 0xFFFFu) | ((tensor_h & 0xFFFFu) << 16));
    g1[3] = (int)(((tensor_h >> 16) & 0xFFFFu) | (tile_w << 16));   // tile_dim0
    g1[4] = (int)tile_h;                          // tile_dim1; tile_dim2=0
    g1[5] = (int)stride0;                         // tensor_dim0_stride[31:0]
    g1[6] = 0;                                    // stride0[47:32], dim1_stride lo
    g1[7] = 0;
    v4i z4 = {0, 0, 0, 0};
    v8i z8 = {0, 0, 0, 0, 0, 0, 0, 0};
    __builtin_amdgcn_tensor_load_to_lds(g0, g1, z4, z4, z8, 0);
}

// ---------------------------------------------------------------------------
// style[b,c] = sum_d w[b,d] * style_w[c,d] * STYLE_SCALE + style_b[c]
// ---------------------------------------------------------------------------
__global__ __launch_bounds__(256) void style_kernel(
    const float* __restrict__ w, const float* __restrict__ style_w,
    const float* __restrict__ style_b, float* __restrict__ style)
{
    int idx = blockIdx.x * 256 + threadIdx.x;          // 16*512
    int b = idx >> 9, c = idx & 511;
    const float4* wr = (const float4*)(w + b * WDIM);
    const float4* sr = (const float4*)(style_w + c * WDIM);
    float acc = 0.f;
    #pragma unroll 8
    for (int j = 0; j < WDIM / 4; ++j) {
        float4 a = wr[j], s = sr[j];
        acc += a.x * s.x + a.y * s.y + a.z * s.z + a.w * s.w;
    }
    style[idx] = acc * STYLE_SCALE + style_b[c];
}

// ---------------------------------------------------------------------------
// wsq[o,i] = sum_r weight[o,i,r]^2          (exact fp32, for demod)
// ---------------------------------------------------------------------------
__global__ __launch_bounds__(256) void wsq_kernel(
    const float* __restrict__ weight, float* __restrict__ wsq)
{
    int idx = blockIdx.x * 256 + threadIdx.x;          // 512*512
    const float* p = weight + idx * 9;
    float s = 0.f;
    #pragma unroll
    for (int j = 0; j < 9; ++j) s += p[j] * p[j];
    wsq[idx] = s;
}

// ---------------------------------------------------------------------------
// d[b,o] = rsqrt(CONV_SCALE^2 * sum_i wsq[o,i]*(style[b,i]+1)^2 + 1e-8)
// ---------------------------------------------------------------------------
__global__ __launch_bounds__(256) void demod_kernel(
    const float* __restrict__ wsq, const float* __restrict__ style,
    float* __restrict__ d)
{
    int idx = blockIdx.x * 256 + threadIdx.x;          // 16*512
    int b = idx >> 9, o = idx & 511;
    const float* wo = wsq + o * IN_C;
    const float* st = style + b * IN_C;
    float s = 0.f;
    #pragma unroll 4
    for (int i = 0; i < IN_C; ++i) {
        float m = st[i] + 1.f;
        s += wo[i] * m * m;
    }
    d[idx] = rsqrtf(s * CONV_SCALE * CONV_SCALE + 1e-8f);
}

// ---------------------------------------------------------------------------
// xm[b,i,h,w] = bf16( x[b,i,h,w] * (style[b,i]+1) )   (modulation on input)
// ---------------------------------------------------------------------------
__global__ __launch_bounds__(256) void modx_kernel(
    const float* __restrict__ x, const float* __restrict__ style,
    unsigned short* __restrict__ xm)
{
    int idx = blockIdx.x * 256 + threadIdx.x;          // 16*512*1024
    int bi = idx >> 10;                                // b*512 + i
    float s = style[bi] + 1.f;
    xm[idx] = f2bf(x[idx] * s);
}

// ---------------------------------------------------------------------------
// wAr[r][o][i] = bf16( weight[o,i,r] * CONV_SCALE )   (K-contiguous A matrix)
// ---------------------------------------------------------------------------
__global__ __launch_bounds__(256) void wprep_kernel(
    const float* __restrict__ weight, unsigned short* __restrict__ wAr)
{
    int idx = blockIdx.x * 256 + threadIdx.x;          // 9*512*512
    int r = idx >> 18;
    int o = (idx >> 9) & 511;
    int i = idx & 511;
    wAr[idx] = f2bf(weight[(o * IN_C + i) * 9 + r] * CONV_SCALE);
}

// ---------------------------------------------------------------------------
// Implicit-GEMM conv via v_wmma_f32_16x16x32_bf16.
// Workgroup = 256 thr = 8 wave32, computes 128 (O) x 128 (pixels) per batch.
// Channel-chunk outer loop (16 x 32ch): stage halo-padded transposed input
// block lX[6][34][32] once, reuse for all 9 taps. A tiles stream through a
// double-buffered LDS tile via the Tensor Data Mover (tensor_load_to_lds),
// overlapped with WMMA and fenced by s_wait_tensorcnt.
// ---------------------------------------------------------------------------
__global__ __launch_bounds__(256) void conv_wmma_kernel(
    const unsigned short* __restrict__ wAr,
    const unsigned short* __restrict__ xm,
    const float* __restrict__ d,
    float* __restrict__ out)
{
    __shared__ __align__(16) unsigned short lA[2][128 * 32];  // 2 x 8 KB
    __shared__ __align__(16) unsigned short lX[LX_ELEMS];     // 12.75 KB

    const int b     = blockIdx.z;
    const int oTile = blockIdx.y * 128;
    const int pTile = blockIdx.x * 128;               // 4 image rows
    const int h0    = pTile >> 5;

    const int t    = threadIdx.x;
    const int lane = t & 31;
    const int wave = t >> 5;
    const int wm   = wave & 1;                        // 0..1 -> 64 rows of O
    const int wn   = wave >> 1;                       // 0..3 -> 32 pixels
    const int lo   = lane & 15;
    const int hi   = lane >> 4;                       // lane-half

    // fragment K offsets per ISA 7.12.2 bf16 layouts
    const int kb_a = hi ? 8 : 0;                      // A: e0-7->K=kb_a.., e8-15->K=kb_a+16..
    const int kb_b = hi ? 8 : 0;                      // B: e0-7->K=kb_b.., e8-15->K=kb_b+16..

    v8f acc[4][2];
    #pragma unroll
    for (int fm = 0; fm < 4; ++fm)
        #pragma unroll
        for (int fn = 0; fn < 2; ++fn)
            acc[fm][fn] = v8f{0.f,0.f,0.f,0.f,0.f,0.f,0.f,0.f};

    const unsigned short* xb = xm + (size_t)b * IN_C * HW;

    // prologue: kick off TDM for (tap 0, chunk 0) into buffer 0
    if (wave == 0) {
        tdm_load_tileA((u32)(uintptr_t)(void*)&lA[0][0],
                       wAr + ((size_t)(0 * OUT_C + oTile) * IN_C + 0));
    }

    int cur = 0;
    for (int i0 = 0; i0 < IN_C; i0 += 32) {
        __syncthreads();     // all taps of previous chunk done -> lX reusable

        // ---- stage halo-padded, transposed input block: lX[pos][k] ----
        for (int idx = t; idx < LX_ELEMS; idx += 256) {
            int k   = idx / 204;                      // 0..31 channel in chunk
            int pos = idx - k * 204;                  // 0..203 = rp*34 + col
            int rp  = pos / 34;                       // padded row 0..5
            int col = pos - rp * 34;                  // padded col 0..33
            int h   = h0 + rp - 1;
            int wc  = col - 1;
            unsigned short v = 0;
            if ((unsigned)h < (unsigned)IMG && (unsigned)wc < (unsigned)IMG)
                v = xb[((size_t)(i0 + k) * IMG + h) * IMG + wc];
            lX[pos * 32 + k] = v;
        }
        // prefetch next chunk's input rows into cache
        if (i0 + 32 < IN_C)
            __builtin_prefetch(xb + ((size_t)(i0 + 32 + (t & 31)) * IMG + h0) * IMG, 0, 1);

        for (int r = 0; r < 9; ++r) {
            __syncthreads();  // prev-tap LDS reads done before TDM overwrite
            if (wave == 0) {
                if (!(r == 8 && i0 + 32 >= IN_C)) {
                    int nr  = (r == 8) ? 0 : r + 1;
                    int ni0 = (r == 8) ? i0 + 32 : i0;
                    tdm_load_tileA((u32)(uintptr_t)(void*)&lA[cur ^ 1][0],
                                   wAr + ((size_t)(nr * OUT_C + oTile) * IN_C + ni0));
                    __builtin_amdgcn_s_wait_tensorcnt(1);  // current tile ready
                } else {
                    __builtin_amdgcn_s_wait_tensorcnt(0);
                }
            }
            __syncthreads();  // publish lA[cur] (+ lX on r==0) to all waves

            const int dy = r / 3 - 1;
            const int dx = r % 3 - 1;

            // ---- B fragments from shifted window of lX ----
            v16bf bfrag[2];
            #pragma unroll
            for (int fn = 0; fn < 2; ++fn) {
                int p  = wn * 32 + fn * 16 + lo;
                int rl = p >> 5;
                int c  = p & 31;
                const bf16_t* pb = (const bf16_t*)(lX +
                    (((rl + dy + 1) * 34) + (c + dx + 1)) * 32 + kb_b);
                v8bf b0 = *(const v8bf*)(pb);
                v8bf b1 = *(const v8bf*)(pb + 16);
                bfrag[fn] = __builtin_shufflevector(b0, b1,
                    0,1,2,3,4,5,6,7,8,9,10,11,12,13,14,15);
            }
            // ---- A fragments + WMMA ----
            const unsigned short* lAc = &lA[cur][0];
            #pragma unroll
            for (int fm = 0; fm < 4; ++fm) {
                const bf16_t* pa = (const bf16_t*)(lAc + (wm * 64 + fm * 16 + lo) * 32);
                v8bf a0 = *(const v8bf*)(pa + kb_a);
                v8bf a1 = *(const v8bf*)(pa + kb_a + 16);
                v16bf afrag = __builtin_shufflevector(a0, a1,
                    0,1,2,3,4,5,6,7,8,9,10,11,12,13,14,15);
                #pragma unroll
                for (int fn = 0; fn < 2; ++fn) {
                    acc[fm][fn] = __builtin_amdgcn_wmma_f32_16x16x32_bf16(
                        false, afrag, false, bfrag[fn],
                        (short)0, acc[fm][fn], false, false);
                }
            }
            cur ^= 1;
        }
    }

    // ---- epilogue: demodulate and store fp32 ----
    const float* db = d + b * OUT_C;
    float* ob = out + (size_t)b * OUT_C * HW;
    #pragma unroll
    for (int fm = 0; fm < 4; ++fm) {
        int obase = oTile + wm * 64 + fm * 16 + hi * 8;  // M = vgpr + 8*(lane>=16)
        #pragma unroll
        for (int v = 0; v < 8; ++v) {
            int og = obase + v;
            float dv = db[og];
            #pragma unroll
            for (int fn = 0; fn < 2; ++fn) {
                int pg = pTile + wn * 32 + fn * 16 + lo; // N = lane&15
                ob[(size_t)og * HW + pg] = acc[fm][fn][v] * dv;
            }
        }
    }
}

// ---------------------------------------------------------------------------
extern "C" void kernel_launch(void* const* d_in, const int* in_sizes, int n_in,
                              void* d_out, int out_size, void* d_ws, size_t ws_size,
                              hipStream_t stream) {
    const float* x       = (const float*)d_in[0];   // (16,512,32,32)
    const float* w       = (const float*)d_in[1];   // (16,512)
    const float* weight  = (const float*)d_in[2];   // (512,512,3,3)
    const float* style_w = (const float*)d_in[3];   // (512,512)
    const float* style_b = (const float*)d_in[4];   // (512,)
    float* out = (float*)d_out;

    char* ws = (char*)d_ws;
    float*          style = (float*)(ws + 0);                  //  32 KB
    float*          dmod  = (float*)(ws + 32768);              //  32 KB
    float*          wsq   = (float*)(ws + 65536);              //   1 MB
    unsigned short* wAr   = (unsigned short*)(ws + 1114112);   // 4.5 MB
    unsigned short* xm    = (unsigned short*)(ws + 5832704);   //  16 MB

    style_kernel<<<(BATCH * IN_C) / 256, 256, 0, stream>>>(w, style_w, style_b, style);
    wsq_kernel  <<<(OUT_C * IN_C) / 256, 256, 0, stream>>>(weight, wsq);
    demod_kernel<<<(BATCH * OUT_C) / 256, 256, 0, stream>>>(wsq, style, dmod);
    modx_kernel <<<(BATCH * IN_C * HW) / 256, 256, 0, stream>>>(x, style, xm);
    wprep_kernel<<<(9 * OUT_C * IN_C) / 256, 256, 0, stream>>>(weight, wAr);

    dim3 grid(HW / 128, OUT_C / 128, BATCH);   // 8 x 4 x 16
    conv_wmma_kernel<<<grid, 256, 0, stream>>>(wAr, xm, dmod, out);
}
